// MoEImage_26061861552659
// MI455X (gfx1250) — matrix operand order repaired and checked by reading, not compiled, not run on hardware
//
#include <hip/hip_runtime.h>
#include <hip/hip_bf16.h>
#include <math.h>

typedef __attribute__((ext_vector_type(16))) __bf16 v16bf;
typedef __attribute__((ext_vector_type(8)))  float  v8f;
typedef int v4i __attribute__((vector_size(16)));           // matches builtin sig

#define BATCH 32
#define CH 64            // CIN == HID == OUT
#define NPIX 16384       // 128*128
#define TILE_P 256       // pixels per block
#define FSTRIDE 72       // padded LDS row stride for x staging (bf16 elems)
#define NTILES_PER_B (NPIX / TILE_P)   // 64
#define TILE_HW 16384    // halfwords per feature tile: 256 pix * 64 ch

// ---- gfx1250 feature probes (degrade gracefully if builtins absent) -------
#if defined(__has_builtin)
#  if __has_builtin(__builtin_amdgcn_global_load_async_to_lds_b128) && \
      __has_builtin(__builtin_amdgcn_global_store_async_from_lds_b128) && \
      __has_builtin(__builtin_amdgcn_s_wait_asynccnt)
#    define USE_ASYNC_LDS 1
#  endif
#  if __has_builtin(__builtin_amdgcn_tanhf)
#    define HAVE_HW_TANH 1
#  endif
#endif
#ifndef USE_ASYNC_LDS
#  define USE_ASYNC_LDS 0
#endif
#ifndef HAVE_HW_TANH
#  define HAVE_HW_TANH 0
#endif

#if USE_ASYNC_LDS
typedef __attribute__((address_space(1))) v4i* gptr_v4i;   // global (prints as __device__)
typedef __attribute__((address_space(3))) v4i* lptr_v4i;   // LDS
#endif

__device__ __forceinline__ unsigned short f2bfu(float f) {
  unsigned u = __builtin_bit_cast(unsigned, f);
  unsigned r = u + 0x7FFFu + ((u >> 16) & 1u);   // round-to-nearest-even
  return (unsigned short)(r >> 16);
}

// branch-free tanh-form GELU (v_tanh_f32 on gfx1250 if available)
__device__ __forceinline__ float gelu_f(float x) {
  float u = 0.7978845608028654f * x * (1.0f + 0.044715f * x * x);
#if HAVE_HW_TANH
  float t = __builtin_amdgcn_tanhf(u);
#else
  float t = 1.0f - 2.0f / (__expf(2.0f * u) + 1.0f);
#endif
  return 0.5f * x * (1.0f + t);
}

union FragAB {
  v16bf v;
  uint4 q[2];
};

// Feature tile layout in global ws (per 256-pixel tile): 1024 chunks of 32B.
// chunk index = (nt*2 + kt)*32 + lane ; chunk = one lane's v16bf B-fragment.
// This makes k3's staging a contiguous 32KB copy and frag reads conflict-free.

// ---------------------------------------------------------------------------
// Kernel 1: features = GELU(fe_w @ x + fe_b); store in B-fragment order (bf16)
// and accumulate per-(b,c) sums for the gating global-average-pool.
// ---------------------------------------------------------------------------
__global__ __launch_bounds__(256) void k_features(
    const float* __restrict__ x, const float* __restrict__ fe_w,
    const float* __restrict__ fe_b,
    unsigned short* __restrict__ feat, float* __restrict__ gf)
{
  __shared__ __align__(16) unsigned short sA[CH * CH];          // 8 KB weights
  __shared__ __align__(16) unsigned short sF[TILE_P * FSTRIDE]; // 36 KB x tile
  __shared__ __align__(16) unsigned short sFrag[TILE_HW];       // 32 KB result
  __shared__ float sBias[CH];
  __shared__ float sGF[CH];

  const int tid  = threadIdx.x;
  const int b    = blockIdx.x / NTILES_PER_B;
  const int tile = blockIdx.x % NTILES_PER_B;
  const int p0   = tile * TILE_P;

  // stage fe_w fp32 -> bf16, row-major [out][in]
  for (int i = tid * 4; i < CH * CH; i += 256 * 4) {
    float4 w4 = *(const float4*)(fe_w + i);
    sA[i + 0] = f2bfu(w4.x); sA[i + 1] = f2bfu(w4.y);
    sA[i + 2] = f2bfu(w4.z); sA[i + 3] = f2bfu(w4.w);
  }
  if (tid < CH) { sBias[tid] = fe_b[tid]; sGF[tid] = 0.0f; }

  // stage x tile transposed into LDS: sF[p][c] (c contiguous), fp32 -> bf16
  {
    const float* xp = x + ((size_t)b * CH) * NPIX + (p0 + tid);
    unsigned* dst = (unsigned*)&sF[tid * FSTRIDE];
#pragma unroll
    for (int c = 0; c < CH; c += 2) {
      float v0 = xp[(size_t)c * NPIX];
      float v1 = xp[(size_t)(c + 1) * NPIX];
      dst[c >> 1] = (unsigned)f2bfu(v0) | ((unsigned)f2bfu(v1) << 16);
    }
    __builtin_prefetch(xp + TILE_P, 0, 0);   // next tile -> global_prefetch_b8
  }
  __syncthreads();

  const int wave = tid >> 5, lane = tid & 31;
  const int mt = wave & 3, ntBase = (wave >> 2) * 8;
  const int m0 = mt * 16;
  const int mrow = m0 + (lane & 15);
  const int hi = lane >> 4;              // lane half
  const int kh = hi ? 8 : 0;             // 16-bit A 16x32 lane K layout
  const int kB = hi ? 8 : 0;             // 16-bit B 32x16 lane K layout

  FragAB a0, a1;
  a0.q[0] = *(const uint4*)&sA[mrow * CH + 0  + kh];
  a0.q[1] = *(const uint4*)&sA[mrow * CH + 16 + kh];
  a1.q[0] = *(const uint4*)&sA[mrow * CH + 32 + kh];
  a1.q[1] = *(const uint4*)&sA[mrow * CH + 48 + kh];

  float biasr[8];
#pragma unroll
  for (int r = 0; r < 8; ++r) biasr[r] = sBias[m0 + r + hi * 8];

  float rowsum[8];
#pragma unroll
  for (int r = 0; r < 8; ++r) rowsum[r] = 0.0f;

  const int pl = lane & 15;              // pixel within nt

  for (int j = 0; j < 8; ++j) {
    const int nt = ntBase + j;
    const int prow = nt * 16 + pl;
    FragAB b0, b1;
    b0.q[0] = *(const uint4*)&sF[prow * FSTRIDE + kB + 0];
    b0.q[1] = *(const uint4*)&sF[prow * FSTRIDE + kB + 16];
    b1.q[0] = *(const uint4*)&sF[prow * FSTRIDE + 32 + kB + 0];
    b1.q[1] = *(const uint4*)&sF[prow * FSTRIDE + 32 + kB + 16];

    v8f acc;
#pragma unroll
    for (int r = 0; r < 8; ++r) acc[r] = biasr[r];

    acc = __builtin_amdgcn_wmma_f32_16x16x32_bf16(false, a0.v, false, b0.v,
                                                  (short)0, acc, false, false);
    acc = __builtin_amdgcn_wmma_f32_16x16x32_bf16(false, a1.v, false, b1.v,
                                                  (short)0, acc, false, false);

#pragma unroll
    for (int r = 0; r < 8; ++r) {
      float g = gelu_f(acc[r]);
      rowsum[r] += g;
      // scatter into B-fragment order: element (K=c, N=p)
      const int c  = m0 + r + hi * 8;
      const int kt = c >> 5;
      const int cr = c & 31;
      const int lane2 = pl + (((cr >> 3) & 1) ? 16 : 0);
      sFrag[(((nt * 2 + kt) * 32 + lane2) << 4) + (((cr >> 4) << 3) | (cr & 7))] = f2bfu(g);
    }
  }

  // one cross-lane reduction for the pooled sums (16 lanes share a row)
#pragma unroll
  for (int r = 0; r < 8; ++r) {
    float s = rowsum[r];
    s += __shfl_xor(s, 1, 32);
    s += __shfl_xor(s, 2, 32);
    s += __shfl_xor(s, 4, 32);
    s += __shfl_xor(s, 8, 32);
    if (pl == 0) atomicAdd(&sGF[m0 + r + hi * 8], s);
  }
  __syncthreads();
  if (tid < CH) atomicAdd(&gf[b * CH + tid], sGF[tid]);

  // drain the 32KB tile to global, contiguous & coalesced
  unsigned short* gdst = feat + (size_t)blockIdx.x * TILE_HW;
#if USE_ASYNC_LDS
#pragma unroll
  for (int i = 0; i < 8; ++i) {
    const int hw = (i * 256 + tid) * 8;
    __builtin_amdgcn_global_store_async_from_lds_b128(
        (gptr_v4i)(gdst + hw), (lptr_v4i)(&sFrag[hw]), 0, 0);
  }
  __builtin_amdgcn_s_wait_asynccnt(0);
#else
#pragma unroll
  for (int i = 0; i < 8; ++i) {
    const int hw = (i * 256 + tid) * 8;
    *(uint4*)(gdst + hw) = *(const uint4*)&sFrag[hw];
  }
#endif
}

// ---------------------------------------------------------------------------
// Kernel 2: gating MLP + channel-attention + top-2 softmax(T=2). 1 thread/sample.
// ---------------------------------------------------------------------------
__global__ __launch_bounds__(32) void k_gate(
    const float* __restrict__ gf,
    const float* __restrict__ g1_w, const float* __restrict__ g1_b,
    const float* __restrict__ bn1_g, const float* __restrict__ bn1_b,
    const float* __restrict__ ca1_w, const float* __restrict__ ca1_b,
    const float* __restrict__ ca2_w, const float* __restrict__ ca2_b,
    const float* __restrict__ g2_w, const float* __restrict__ g2_b,
    const float* __restrict__ bn2_g, const float* __restrict__ bn2_b,
    const float* __restrict__ g3_w, const float* __restrict__ g3_b,
    int* __restrict__ route_idx, float* __restrict__ route_w)
{
  const int bi = threadIdx.x;
  if (bi >= BATCH) return;
  const float bnscale = rsqrtf(1.0f + 1e-5f);

  float gfv[CH];
  for (int c = 0; c < CH; ++c) gfv[c] = gf[bi * CH + c] * (1.0f / (float)NPIX);

  float h1[128];
  for (int i = 0; i < 128; ++i) {
    float v = g1_b[i];
    for (int c = 0; c < CH; ++c) v += gfv[c] * g1_w[i * CH + c];
    h1[i] = gelu_f(v * (bn1_g[i] * bnscale) + bn1_b[i]);
  }
  float a1[8];
  for (int j = 0; j < 8; ++j) {
    float v = ca1_b[j];
    for (int r = 0; r < 128; ++r) v += h1[r] * ca1_w[j * 128 + r];
    a1[j] = gelu_f(v);
  }
  for (int i = 0; i < 128; ++i) {
    float v = ca2_b[i];
    for (int j = 0; j < 8; ++j) v += a1[j] * ca2_w[i * 8 + j];
    h1[i] *= 1.0f / (1.0f + __expf(-2.0f * v));   // sigmoid(att+att)
  }
  float hh[CH];
  for (int o = 0; o < CH; ++o) {
    float v = g2_b[o];
    for (int r = 0; r < 128; ++r) v += h1[r] * g2_w[o * 128 + r];
    hh[o] = gelu_f(v * (bn2_g[o] * bnscale) + bn2_b[o]);
  }
  float sc[8];
  for (int e = 0; e < 8; ++e) {
    float v = g3_b[e];
    for (int c = 0; c < CH; ++c) v += hh[c] * g3_w[e * CH + c];
    sc[e] = v;
  }
  int i0 = 0; float v0 = sc[0];
  for (int e = 1; e < 8; ++e) if (sc[e] > v0) { v0 = sc[e]; i0 = e; }
  int i1 = (i0 == 0) ? 1 : 0; float v1 = sc[i1];
  for (int e = 0; e < 8; ++e) if (e != i0 && sc[e] > v1) { v1 = sc[e]; i1 = e; }
  float m  = fmaxf(v0, v1);
  float e0 = __expf((v0 - m) * 0.5f);   // TEMP = 2.0
  float e1 = __expf((v1 - m) * 0.5f);
  float inv = 1.0f / (e0 + e1);
  route_idx[bi * 2 + 0] = i0;  route_idx[bi * 2 + 1] = i1;
  route_w[bi * 2 + 0] = e0 * inv;  route_w[bi * 2 + 1] = e1 * inv;
}

// ---------------------------------------------------------------------------
// Kernel 3: out = 0.5*(GELU(S0 f)+GELU(S1 f)) + w0*GELU(E_i0 f) + w1*GELU(E_i1 f)
// Feature tile arrives in B-fragment order -> async bulk copy to LDS.
// ---------------------------------------------------------------------------
__global__ __launch_bounds__(256) void k_experts(
    const unsigned short* __restrict__ feat,
    const float* __restrict__ s_w, const float* __restrict__ s_b,
    const float* __restrict__ e_w, const float* __restrict__ e_b,
    const int* __restrict__ route_idx, const float* __restrict__ route_w,
    float* __restrict__ out)
{
  __shared__ __align__(16) unsigned short sW[4 * CH * CH];  // 32 KB weights
  __shared__ __align__(16) unsigned short sFrag[TILE_HW];   // 32 KB features
  __shared__ float sBias[4][CH];

  const int tid  = threadIdx.x;
  const int b    = blockIdx.x / NTILES_PER_B;
  const int tile = blockIdx.x % NTILES_PER_B;

  // kick off the feature-tile copy first (async global -> LDS if available)
  const unsigned short* gsrc = feat + (size_t)blockIdx.x * TILE_HW;
#if USE_ASYNC_LDS
#pragma unroll
  for (int i = 0; i < 8; ++i) {
    const int hw = (i * 256 + tid) * 8;
    __builtin_amdgcn_global_load_async_to_lds_b128(
        (gptr_v4i)(gsrc + hw), (lptr_v4i)(&sFrag[hw]), 0, 0);
  }
#else
#pragma unroll
  for (int i = 0; i < 8; ++i) {
    const int hw = (i * 256 + tid) * 8;
    *(uint4*)&sFrag[hw] = *(const uint4*)(gsrc + hw);
  }
#endif
  __builtin_prefetch(gsrc + TILE_HW, 0, 0);  // next tile

  const int   i0 = route_idx[b * 2 + 0];
  const int   i1 = route_idx[b * 2 + 1];
  const float w0 = route_w[b * 2 + 0];
  const float w1 = route_w[b * 2 + 1];

  const float* srcs[4] = { s_w, s_w + CH * CH,
                           e_w + (size_t)i0 * CH * CH,
                           e_w + (size_t)i1 * CH * CH };
  for (int m = 0; m < 4; ++m) {
    const float* src = srcs[m];
    unsigned short* dst = sW + m * CH * CH;
    for (int i = tid * 4; i < CH * CH; i += 256 * 4) {
      float4 w4 = *(const float4*)(src + i);
      dst[i + 0] = f2bfu(w4.x); dst[i + 1] = f2bfu(w4.y);
      dst[i + 2] = f2bfu(w4.z); dst[i + 3] = f2bfu(w4.w);
    }
  }
  if (tid < CH) {
    sBias[0][tid] = s_b[tid];
    sBias[1][tid] = s_b[CH + tid];
    sBias[2][tid] = e_b[i0 * CH + tid];
    sBias[3][tid] = e_b[i1 * CH + tid];
  }
#if USE_ASYNC_LDS
  __builtin_amdgcn_s_wait_asynccnt(0);
#endif
  __syncthreads();

  const int wave = tid >> 5, lane = tid & 31;
  const int mt = wave & 3, ntBase = (wave >> 2) * 8;
  const int m0 = mt * 16;
  const int mrow = m0 + (lane & 15);
  const int hi = lane >> 4;
  const int kh = hi ? 8 : 0;
  const int pl = lane & 15;

  FragAB A[4][2];
#pragma unroll
  for (int m = 0; m < 4; ++m) {
    const unsigned short* wm = sW + m * CH * CH + mrow * CH;
    A[m][0].q[0] = *(const uint4*)&wm[0  + kh];
    A[m][0].q[1] = *(const uint4*)&wm[16 + kh];
    A[m][1].q[0] = *(const uint4*)&wm[32 + kh];
    A[m][1].q[1] = *(const uint4*)&wm[48 + kh];
  }
  float biasr[4][8];
#pragma unroll
  for (int m = 0; m < 4; ++m)
#pragma unroll
    for (int r = 0; r < 8; ++r) biasr[m][r] = sBias[m][m0 + r + hi * 8];

  const int p0 = tile * TILE_P;

  for (int j = 0; j < 8; ++j) {
    const int nt = ntBase + j;
    // conflict-free fragment reads: lane-consecutive 32B chunks
    FragAB b0, b1;
    {
      const uint4* c0 = (const uint4*)&sFrag[((nt * 2 + 0) * 32 + lane) << 4];
      const uint4* c1 = (const uint4*)&sFrag[((nt * 2 + 1) * 32 + lane) << 4];
      b0.q[0] = c0[0]; b0.q[1] = c0[1];
      b1.q[0] = c1[0]; b1.q[1] = c1[1];
    }

    v8f acc[4];
#pragma unroll
    for (int m = 0; m < 4; ++m) {
#pragma unroll
      for (int r = 0; r < 8; ++r) acc[m][r] = biasr[m][r];
      acc[m] = __builtin_amdgcn_wmma_f32_16x16x32_bf16(
          false, A[m][0].v, false, b0.v, (short)0, acc[m], false, false);
      acc[m] = __builtin_amdgcn_wmma_f32_16x16x32_bf16(
          false, A[m][1].v, false, b1.v, (short)0, acc[m], false, false);
    }

    const int pcol = p0 + nt * 16 + pl;
#pragma unroll
    for (int r = 0; r < 8; ++r) {
      const int c = m0 + r + hi * 8;
      float val = 0.5f * (gelu_f(acc[0][r]) + gelu_f(acc[1][r]))
                + w0 * gelu_f(acc[2][r]) + w1 * gelu_f(acc[3][r]);
      out[((size_t)b * CH + c) * NPIX + pcol] = val;
    }
  }
}

// ---------------------------------------------------------------------------
extern "C" void kernel_launch(void* const* d_in, const int* in_sizes, int n_in,
                              void* d_out, int out_size, void* d_ws, size_t ws_size,
                              hipStream_t stream) {
  const float* x     = (const float*)d_in[0];
  const float* fe_w  = (const float*)d_in[1];
  const float* fe_b  = (const float*)d_in[2];
  const float* s_w   = (const float*)d_in[3];
  const float* s_b   = (const float*)d_in[4];
  const float* e_w   = (const float*)d_in[5];
  const float* e_b   = (const float*)d_in[6];
  const float* g1_w  = (const float*)d_in[7];
  const float* g1_b  = (const float*)d_in[8];
  const float* bn1_g = (const float*)d_in[9];
  const float* bn1_b = (const float*)d_in[10];
  const float* ca1_w = (const float*)d_in[11];
  const float* ca1_b = (const float*)d_in[12];
  const float* ca2_w = (const float*)d_in[13];
  const float* ca2_b = (const float*)d_in[14];
  const float* g2_w  = (const float*)d_in[15];
  const float* g2_b  = (const float*)d_in[16];
  const float* bn2_g = (const float*)d_in[17];
  const float* bn2_b = (const float*)d_in[18];
  const float* g3_w  = (const float*)d_in[19];
  const float* g3_b  = (const float*)d_in[20];

  char* ws = (char*)d_ws;
  unsigned short* feat = (unsigned short*)ws;                        // 64 MB bf16
  const size_t featBytes = (size_t)BATCH * NTILES_PER_B * TILE_HW * sizeof(unsigned short);
  float* gf       = (float*)(ws + featBytes);                        // 8 KB
  int*   route_ix = (int*)(ws + featBytes + BATCH * CH * sizeof(float));
  float* route_w  = (float*)((char*)route_ix + BATCH * 2 * sizeof(int));

  (void)hipMemsetAsync(gf, 0, BATCH * CH * sizeof(float), stream);

  dim3 grid(BATCH * NTILES_PER_B);   // 2048 blocks
  k_features<<<grid, 256, 0, stream>>>(x, fe_w, fe_b, feat, gf);
  k_gate<<<1, 32, 0, stream>>>(gf, g1_w, g1_b, bn1_g, bn1_b, ca1_w, ca1_b,
                               ca2_w, ca2_b, g2_w, g2_b, bn2_g, bn2_b,
                               g3_w, g3_b, route_ix, route_w);
  k_experts<<<grid, 256, 0, stream>>>(feat, s_w, s_b, e_w, e_b,
                                      route_ix, route_w, (float*)d_out);
}
